// GCN_BiLSTM_Attn_Temporal_38431367364697
// MI455X (gfx1250) — matrix operand order repaired and checked by reading, not compiled, not run on hardware
//
#include <hip/hip_runtime.h>
#include <hip/hip_bf16.h>
#include <math.h>

// ---------------- problem constants (match reference) ----------------
#define NN   50000     // nodes  (50000 = 3125 * 16 = 625 * 80)
#define EE   800000    // edges
#define EMB  128
#define GH   64
#define TD   4
#define BB   64        // batch
#define TT   200       // seq len
#define HH   128       // lstm hidden per dir
#define IND  134       // 2 + EMB + TD
#define G4   512       // 4*HH
#define FCN  80        // fc columns per block (50000 / 80 = 625)

typedef __attribute__((ext_vector_type(16))) _Float16 v16h;
typedef __attribute__((ext_vector_type(8)))  float    v8f;

__device__ __forceinline__ v8f wmma16(v16h a, v16h b, v8f c) {
  // D = A(16x32 f16) * B(32x16 f16) + C(16x16 f32)
  return __builtin_amdgcn_wmma_f32_16x16x32_f16(false, a, false, b, (short)0, c, false, false);
}
__device__ __forceinline__ float sigm(float x) { return 1.0f / (1.0f + expf(-x)); }

// A-layout k index for element j, lane L:  k = 16*(j>>3) + 8*(L>>4) + (j&7)
#define AK(j, half) (16 * ((j) >> 3) + 8 * (half) + ((j) & 7))

// ---------------- utility kernels ----------------
__global__ void k_fill(float* p, float v, size_t n) {
  size_t i = (size_t)blockIdx.x * blockDim.x + threadIdx.x;
  if (i < n) p[i] = v;
}

// deg[dst] += ew  (deg pre-filled with 1.0 for the self loop)
__global__ void k_deg(const int* ei, const float* ew, float* deg) {
  int e = blockIdx.x * blockDim.x + threadIdx.x;
  if (e < EE) atomicAdd(&deg[ei[EE + e]], ew[e]);
}
__global__ void k_dinv(float* d) {
  int i = blockIdx.x * blockDim.x + threadIdx.x;
  if (i < NN) { float v = d[i]; d[i] = (v > 0.f) ? rsqrtf(v) : 0.f; }
}

// ---------------- GCN layer 1 linear: [N,134] @ [134,64] via WMMA ----------------
__global__ void __launch_bounds__(128) k_gcn1(const float* __restrict__ xc,
                                              const float* __restrict__ tf,
                                              const float* __restrict__ emb,
                                              const float* __restrict__ W,
                                              float* __restrict__ h1) {
  int tile = blockIdx.x;                 // 3125 row tiles
  int wave = threadIdx.x >> 5;           // 4 waves = 4 col tiles (GH=64)
  int lane = threadIdx.x & 31;
  int m = lane & 15, half = lane >> 4;
  int row = tile * 16 + m;               // A row for this lane
  int n0 = wave * 16;
  v8f acc = {};
  for (int kc = 0; kc < 5; ++kc) {       // K padded 134 -> 160
    int k0 = kc * 32;
    v16h a, b;
#pragma unroll
    for (int j = 0; j < 16; ++j) {
      int k = k0 + AK(j, half);
      float v = 0.f;
      if (k < 2)        v = xc[row * 2 + k];
      else if (k < 130) v = emb[row * 128 + (k - 2)];
      else if (k < 134) v = tf[row * 4 + (k - 130)];
      a[j] = (_Float16)v;
    }
    int nb = n0 + (lane & 15);           // B: n = lane % 16, k = AK(j, half)
#pragma unroll
    for (int j = 0; j < 16; ++j) {
      int kb = k0 + AK(j, half);
      float v = (kb < IND) ? W[kb * GH + nb] : 0.f;
      b[j] = (_Float16)v;
    }
    acc = wmma16(a, b, acc);
  }
  int n = n0 + (lane & 15);
#pragma unroll
  for (int r = 0; r < 8; ++r)
    h1[(size_t)(tile * 16 + r + 8 * half) * GH + n] = acc[r];
}

// ---------------- edge scatter-add: out[dst] += h[src]*norm ----------------
__global__ void k_agg(const int* __restrict__ ei, const float* __restrict__ ew,
                      const float* __restrict__ dinv, const float* __restrict__ h,
                      float* __restrict__ out, int F) {
  int e = blockIdx.x;
  int f = threadIdx.x;
  int s = ei[e], d = ei[EE + e];
  float nrm = dinv[s] * ew[e] * dinv[d];
  atomicAdd(&out[(size_t)d * F + f], h[(size_t)s * F + f] * nrm);
}

// add self-loop term + bias (+optional relu)
__global__ void k_fin(float* __restrict__ out, const float* __restrict__ h,
                      const float* __restrict__ dinv, const float* __restrict__ b,
                      int F, int relu) {
  int v = blockIdx.x, f = threadIdx.x;
  float di = dinv[v];
  float r = out[(size_t)v * F + f] + h[(size_t)v * F + f] * di * di + b[f];
  if (relu) r = fmaxf(r, 0.f);
  out[(size_t)v * F + f] = r;
}

// ---------------- GCN layer 2 linear: [N,64] @ [64,128] ----------------
__global__ void __launch_bounds__(256) k_gcn2(const float* __restrict__ z1,
                                              const float* __restrict__ W,
                                              float* __restrict__ h2) {
  int tile = blockIdx.x;                 // 3125
  int wave = threadIdx.x >> 5;           // 8 waves = 8 col tiles (128)
  int lane = threadIdx.x & 31;
  int m = lane & 15, half = lane >> 4;
  int row = tile * 16 + m;
  int n0 = wave * 16;
  v8f acc = {};
  for (int kc = 0; kc < 2; ++kc) {       // K = 64
    int k0 = kc * 32;
    v16h a, b;
#pragma unroll
    for (int j = 0; j < 16; ++j) a[j] = (_Float16)z1[(size_t)row * GH + k0 + AK(j, half)];
    int nb = n0 + (lane & 15);
#pragma unroll
    for (int j = 0; j < 16; ++j) b[j] = (_Float16)W[(k0 + AK(j, half)) * EMB + nb];
    acc = wmma16(a, b, acc);
  }
  int n = n0 + (lane & 15);
#pragma unroll
  for (int r = 0; r < 8; ++r)
    h2[(size_t)(tile * 16 + r + 8 * half) * EMB + n] = acc[r];
}

// ---------------- gather seq embeddings (fwd + length-reversed) -> f16 [2][T][B][128]
__global__ void k_gather(const float* __restrict__ z, const int* __restrict__ seq,
                         const int* __restrict__ len, _Float16* __restrict__ semb) {
  int t = blockIdx.x, b = blockIdx.y, dir = blockIdx.z, u = threadIdx.x;
  int L = len[b];
  int tt = (dir == 0) ? t : ((t < L) ? (L - 1 - t) : t);
  int node = seq[b * TT + tt];
  semb[(((size_t)dir * TT + t) * BB + b) * EMB + u] = (_Float16)z[(size_t)node * EMB + u];
}

// ---------------- transpose LSTM weights to f16 [4][128][512]; bsum [2][512] ----------------
__global__ void k_prepw(const float* wif, const float* whf, const float* wib, const float* whb,
                        const float* bif, const float* bhf, const float* bib, const float* bhb,
                        _Float16* __restrict__ wT, float* __restrict__ bsum) {
  int i = blockIdx.x * blockDim.x + threadIdx.x;     // 4*128*512 = 262144
  if (i < 4 * 128 * G4) {
    int w = i >> 16, rem = i & 65535, row = rem >> 9, col = rem & 511;
    const float* src = (w == 0) ? wif : (w == 1) ? whf : (w == 2) ? wib : whb;
    wT[i] = (_Float16)src[col * 128 + row];          // [512,128] -> [128,512]
  }
  if (i < 2 * G4) {
    int d = i >> 9, n = i & 511;
    bsum[i] = d ? (bib[n] + bhb[n]) : (bif[n] + bhf[n]);
  }
}

// ---------------- input projection: [T*B,128] @ w_ih^T[128,512] + bias ----------------
__global__ void __launch_bounds__(256) k_xpre(const _Float16* __restrict__ semb,
                                              const _Float16* __restrict__ wT,
                                              const float* __restrict__ bsum,
                                              float* __restrict__ xpre) {
  int tile = blockIdx.x;                 // 800 row tiles over T*B
  int dir = blockIdx.y;
  int wave = threadIdx.x >> 5;           // 8 waves; each does 4 of 32 N tiles
  int lane = threadIdx.x & 31;
  int m = lane & 15, half = lane >> 4;
  const _Float16* A  = semb + (size_t)dir * TT * BB * EMB;
  const _Float16* Bw = wT + (size_t)dir * 2 * (128 * G4);   // w_ih of this dir
  int row = tile * 16 + m;
  for (int nti = 0; nti < 4; ++nti) {
    int n0 = (wave + 8 * nti) * 16;
    v8f acc = {};
    for (int kc = 0; kc < 4; ++kc) {     // K = 128
      int k0 = kc * 32;
      v16h a, b;
#pragma unroll
      for (int j = 0; j < 16; ++j) a[j] = A[(size_t)row * EMB + k0 + AK(j, half)];
#pragma unroll
      for (int j = 0; j < 16; ++j) b[j] = Bw[(size_t)(k0 + AK(j, half)) * G4 + n0 + (lane & 15)];
      acc = wmma16(a, b, acc);
    }
    int n = n0 + (lane & 15);
#pragma unroll
    for (int r = 0; r < 8; ++r)
      xpre[((size_t)dir * TT * BB + tile * 16 + r + 8 * half) * G4 + n] = acc[r] + bsum[dir * G4 + n];
  }
}

// ---------------- persistent recurrence: grid(4 batch-tiles, 2 dirs), 8 waves/WG ----------------
// Each wave owns a gate quartet (its 16 hidden units x 16 sequences); c-state in regs, h in LDS.
__global__ void __launch_bounds__(256) k_lstm(const _Float16* __restrict__ wT,
                                              const float* __restrict__ xpre,
                                              float* __restrict__ hscan) {
  int mt = blockIdx.x;                   // batch tile (16 sequences)
  int dir = blockIdx.y;
  int wave = threadIdx.x >> 5;           // ut = hidden tile 0..7
  int lane = threadIdx.x & 31;
  int m = lane & 15, half = lane >> 4, nc = lane & 15;
  const _Float16* whhT = wT + ((size_t)dir * 2 + 1) * (128 * G4);
  __shared__ _Float16 sh[16 * HH];       // h_{t-1} for this batch tile, f16
  for (int i = threadIdx.x; i < 16 * HH; i += 256) sh[i] = (_Float16)0.f;
  __syncthreads();
  float cst[8];
#pragma unroll
  for (int r = 0; r < 8; ++r) cst[r] = 0.f;
  int ut = wave;
  for (int t = 0; t < TT; ++t) {
    const float* xp = xpre + ((size_t)dir * TT + t) * BB * G4;
    v8f acc[4];
#pragma unroll
    for (int g = 0; g < 4; ++g) {
      int n0 = g * HH + ut * 16;
#pragma unroll
      for (int r = 0; r < 8; ++r)
        acc[g][r] = xp[(size_t)(mt * 16 + r + 8 * half) * G4 + n0 + nc];
    }
#pragma unroll
    for (int kc = 0; kc < 4; ++kc) {     // K = 128 over h
      int k0 = kc * 32;
      v16h a;
#pragma unroll
      for (int j = 0; j < 16; ++j) a[j] = sh[m * HH + k0 + AK(j, half)];
#pragma unroll
      for (int g = 0; g < 4; ++g) {
        int n0 = g * HH + ut * 16;
        v16h b;
#pragma unroll
        for (int j = 0; j < 16; ++j) b[j] = whhT[(size_t)(k0 + AK(j, half)) * G4 + n0 + nc];
        acc[g] = wmma16(a, b, acc[g]);
      }
    }
    float hnew[8];
#pragma unroll
    for (int r = 0; r < 8; ++r) {
      float ig = sigm(acc[0][r]);
      float fg = sigm(acc[1][r]);
      float gg = tanhf(acc[2][r]);
      float og = sigm(acc[3][r]);
      float c = fg * cst[r] + ig * gg;
      cst[r] = c;
      hnew[r] = og * tanhf(c);
    }
    __syncthreads();                     // all reads of sh (h_{t-1}) done
#pragma unroll
    for (int r = 0; r < 8; ++r) {
      int lb = r + 8 * half;             // local sequence row 0..15
      int u = ut * 16 + nc;              // hidden unit
      sh[lb * HH + u] = (_Float16)hnew[r];
      hscan[(((size_t)dir * TT + t) * BB + mt * 16 + lb) * HH + u] = hnew[r];
    }
    __syncthreads();                     // h_t visible for next step
  }
}

// ---------------- masked attention pooling -> context [64][256] ----------------
__global__ void __launch_bounds__(256) k_attn(const float* __restrict__ hscan,
                                              const int* __restrict__ len,
                                              const float* __restrict__ aw,
                                              const float* __restrict__ ab,
                                              float* __restrict__ ctx,
                                              _Float16* __restrict__ ctxh) {
  int b = blockIdx.x, tid = threadIdx.x;
  __shared__ float sc[TT];
  __shared__ float red[256];
  int L = len[b];
  const float NEG = -1e30f;
  for (int t = tid; t < TT; t += 256) {
    float s = ab[0];
    const float* hf = &hscan[((size_t)t * BB + b) * HH];
    for (int d = 0; d < HH; ++d) s += hf[d] * aw[d];
    int tt = (t < L) ? (L - 1 - t) : t;
    const float* hb = &hscan[(((size_t)TT + tt) * BB + b) * HH];
    for (int d = 0; d < HH; ++d) s += hb[d] * aw[HH + d];
    sc[t] = (t < L) ? s : NEG;
  }
  __syncthreads();
  float mx = NEG;
  for (int t = tid; t < TT; t += 256) mx = fmaxf(mx, sc[t]);
  red[tid] = mx; __syncthreads();
  for (int s = 128; s > 0; s >>= 1) { if (tid < s) red[tid] = fmaxf(red[tid], red[tid + s]); __syncthreads(); }
  mx = red[0]; __syncthreads();
  float sum = 0.f;
  for (int t = tid; t < TT; t += 256) { float e = expf(sc[t] - mx); sc[t] = e; sum += e; }
  red[tid] = sum; __syncthreads();
  for (int s = 128; s > 0; s >>= 1) { if (tid < s) red[tid] += red[tid + s]; __syncthreads(); }
  float inv = 1.f / red[0]; __syncthreads();
  int d = tid;                           // 256 dims, 256 threads
  float acc = 0.f;
  for (int t = 0; t < TT; ++t) {
    float v;
    if (d < HH) v = hscan[((size_t)t * BB + b) * HH + d];
    else { int tt = (t < L) ? (L - 1 - t) : t; v = hscan[(((size_t)TT + tt) * BB + b) * HH + (d - HH)]; }
    acc += sc[t] * inv * v;
  }
  ctx[b * 256 + d] = acc;
  ctxh[b * 256 + d] = (_Float16)acc;
}

// ---------------- FC: [64,256] @ fc_w[256,50000] + fc_b ----------------
// One block = all 64 batch rows x 80 output columns. The K=256 x 80col panel of
// fc_w (80 KB) is staged into LDS ONCE via CDNA5 async-to-LDS loads (ASYNCcnt),
// then 20 waves (4 M-tiles x 5 N-tiles) consume it with WMMA.
__global__ void __launch_bounds__(640) k_fc(const _Float16* __restrict__ ctxh,
                                            const float* __restrict__ fcw,
                                            const float* __restrict__ fcb,
                                            float* __restrict__ out) {
  __shared__ float sB[256 * FCN];        // 80 KB
  int n0 = blockIdx.x * FCN;             // 625 blocks
  int tid = threadIdx.x;

  // --- async stage: 256 rows x 80 floats = 5120 x 16B chunks, 8 per thread ---
  unsigned ldsbase = (unsigned)(size_t)(&sB[0]);   // flat shared addr[31:0] == LDS offset
#pragma unroll
  for (int i = 0; i < 8; ++i) {
    int c = tid + 640 * i;               // chunk id
    int r = c / 20, cc = c % 20;         // row 0..255, 16B chunk 0..19
    unsigned laddr = ldsbase + (unsigned)(r * FCN + cc * 4) * 4u;
    unsigned goff  = (unsigned)((size_t)r * NN * 4u + (size_t)(n0 + cc * 4) * 4u);
    asm volatile("global_load_async_to_lds_b128 %0, %1, %2"
                 :: "v"(laddr), "v"(goff), "s"(fcw) : "memory");
  }
  asm volatile("s_wait_asynccnt 0x0" ::: "memory");
  __syncthreads();

  // --- compute: wave w -> M tile (w/5), N tile (w%5) ---
  int wave = tid >> 5, lane = tid & 31;
  int mt = wave / 5, nt = wave % 5;
  int m = lane & 15, half = lane >> 4;
  v8f acc = {};
  for (int kc = 0; kc < 8; ++kc) {       // K = 256
    int k0 = kc * 32;
    v16h a, b;
#pragma unroll
    for (int j = 0; j < 16; ++j) a[j] = ctxh[(mt * 16 + m) * 256 + k0 + AK(j, half)];
#pragma unroll
    for (int j = 0; j < 16; ++j) b[j] = (_Float16)sB[(k0 + AK(j, half)) * FCN + nt * 16 + (lane & 15)];
    acc = wmma16(a, b, acc);
  }
  int n = n0 + nt * 16 + (lane & 15);
#pragma unroll
  for (int r = 0; r < 8; ++r)
    out[(size_t)(mt * 16 + r + 8 * half) * NN + n] = acc[r] + fcb[n];
}

// ---------------- host-side orchestration ----------------
extern "C" void kernel_launch(void* const* d_in, const int* in_sizes, int n_in,
                              void* d_out, int out_size, void* d_ws, size_t ws_size,
                              hipStream_t stream) {
  const float* x_coords = (const float*)d_in[0];
  const float* temporal = (const float*)d_in[1];
  const int*   edge_idx = (const int*)d_in[2];
  const float* edge_w   = (const float*)d_in[3];
  const int*   seq      = (const int*)d_in[4];
  const int*   lengths  = (const int*)d_in[5];
  const float* node_emb = (const float*)d_in[6];
  const float* gcn1_w   = (const float*)d_in[7];
  const float* gcn1_b   = (const float*)d_in[8];
  const float* gcn2_w   = (const float*)d_in[9];
  const float* gcn2_b   = (const float*)d_in[10];
  const float* w_ih_f   = (const float*)d_in[11];
  const float* w_hh_f   = (const float*)d_in[12];
  const float* b_ih_f   = (const float*)d_in[13];
  const float* b_hh_f   = (const float*)d_in[14];
  const float* w_ih_b   = (const float*)d_in[15];
  const float* w_hh_b   = (const float*)d_in[16];
  const float* b_ih_b   = (const float*)d_in[17];
  const float* b_hh_b   = (const float*)d_in[18];
  const float* attn_w   = (const float*)d_in[19];
  const float* attn_b   = (const float*)d_in[20];
  const float* fc_w     = (const float*)d_in[21];
  const float* fc_b     = (const float*)d_in[22];
  float* out = (float*)d_out;

  char* w = (char*)d_ws;
  size_t off = 0;
  auto alloc = [&](size_t bytes) -> void* {
    off = (off + 255) & ~(size_t)255;
    void* p = w + off; off += bytes; return p;
  };
  float*     dinv  = (float*)alloc((size_t)NN * 4);
  float*     h1    = (float*)alloc((size_t)NN * GH * 4);
  float*     a1    = (float*)alloc((size_t)NN * GH * 4);   // becomes z1
  float*     h2    = (float*)alloc((size_t)NN * EMB * 4);
  float*     a2    = (float*)alloc((size_t)NN * EMB * 4);  // becomes z
  _Float16*  semb  = (_Float16*)alloc((size_t)2 * TT * BB * EMB * 2);
  _Float16*  wT    = (_Float16*)alloc((size_t)4 * 128 * G4 * 2);
  float*     bsum  = (float*)alloc((size_t)2 * G4 * 4);
  float*     xpre  = (float*)alloc((size_t)2 * TT * BB * G4 * 4);
  float*     hscan = (float*)alloc((size_t)2 * TT * BB * HH * 4);
  float*     ctx   = (float*)alloc((size_t)BB * 256 * 4);
  _Float16*  ctxh  = (_Float16*)alloc((size_t)BB * 256 * 2);
  (void)ws_size; (void)n_in; (void)in_sizes; (void)out_size;

  // GCN normalization
  k_fill<<<(NN + 255) / 256, 256, 0, stream>>>(dinv, 1.0f, NN);      // self-loop weight
  k_deg<<<(EE + 255) / 256, 256, 0, stream>>>(edge_idx, edge_w, dinv);
  k_dinv<<<(NN + 255) / 256, 256, 0, stream>>>(dinv);

  // GCN layer 1
  k_gcn1<<<NN / 16, 128, 0, stream>>>(x_coords, temporal, node_emb, gcn1_w, h1);
  k_fill<<<(size_t)(NN * GH + 255) / 256, 256, 0, stream>>>(a1, 0.f, (size_t)NN * GH);
  k_agg<<<EE, GH, 0, stream>>>(edge_idx, edge_w, dinv, h1, a1, GH);
  k_fin<<<NN, GH, 0, stream>>>(a1, h1, dinv, gcn1_b, GH, 1);

  // GCN layer 2
  k_gcn2<<<NN / 16, 256, 0, stream>>>(a1, gcn2_w, h2);
  k_fill<<<(size_t)(NN * EMB + 255) / 256, 256, 0, stream>>>(a2, 0.f, (size_t)NN * EMB);
  k_agg<<<EE, EMB, 0, stream>>>(edge_idx, edge_w, dinv, h2, a2, EMB);
  k_fin<<<NN, EMB, 0, stream>>>(a2, h2, dinv, gcn2_b, EMB, 0);

  // sequence gather (fwd + reversed), LSTM weight prep, input projection
  k_gather<<<dim3(TT, BB, 2), EMB, 0, stream>>>(a2, seq, lengths, semb);
  k_prepw<<<(4 * 128 * G4 + 255) / 256, 256, 0, stream>>>(w_ih_f, w_hh_f, w_ih_b, w_hh_b,
                                                          b_ih_f, b_hh_f, b_ih_b, b_hh_b,
                                                          wT, bsum);
  k_xpre<<<dim3((TT * BB) / 16, 2), 256, 0, stream>>>(semb, wT, bsum, xpre);

  // recurrence (2 dirs x 4 independent batch tiles)
  k_lstm<<<dim3(4, 2), 256, 0, stream>>>(wT, xpre, hscan);

  // attention pooling + FC
  k_attn<<<BB, 256, 0, stream>>>(hscan, lengths, attn_w, attn_b, ctx, ctxh);
  k_fc<<<NN / FCN, 640, 0, stream>>>(ctxh, fc_w, fc_b, out);
}
